// MNL_loss_37168646980390
// MI455X (gfx1250) — compile-verified, matrix operand order checked by reading, not run on hardware
//
#include <hip/hip_runtime.h>

typedef __attribute__((ext_vector_type(16))) _Float16 v16h;
typedef __attribute__((ext_vector_type(8)))  _Float16 v8h;
typedef __attribute__((ext_vector_type(8)))  float    v8f;

#define WAVES   8
#define ROWSPW  16          // rows per wave
#define LDS_K   152         // padded halves per row: 304B (16B aligned, bank-friendly)

// Kernel 1: each wave handles 16 rows.
// Phase A: predicated coalesced gathers (skip masked tail -> ~half the HBM traffic),
//          exp(x) stored to LDS as f16; chosen utility reduced exactly in f32 via
//          wave32 butterfly shuffles.
// Phase B: K=128 row-sum via 4x v_wmma_f32_16x16x32_f16 against a ones B-operand,
//          f32 accumulation; per-row loss = chosen - log(rowsum); block tree-reduce.
__global__ __launch_bounds__(256) void mnl_main_kernel(
    const float* __restrict__ x, const float* __restrict__ y,
    const int* __restrict__ idx, const int* __restrict__ lengths,
    float* __restrict__ ws, int Brows)
{
    __shared__ __attribute__((aligned(16))) _Float16 E[WAVES][ROWSPW][LDS_K];
    __shared__ float Ch[WAVES][ROWSPW];
    __shared__ float sdata[256];

    const int tid  = threadIdx.x;
    const int wave = tid >> 5;
    const int lane = tid & 31;
    const int rowBase = blockIdx.x * (WAVES * ROWSPW) + wave * ROWSPW;

    for (int rr = 0; rr < ROWSPW; ++rr) {
        const int row = rowBase + rr;
        int len = 0;
        if (row < Brows) {
            len = lengths[row];
            if (len < 1)   len = 1;    // reference clips to [1, A]
            if (len > 128) len = 128;
        }
        const long base = (long)row * 128;
        float cacc = 0.0f;
        #pragma unroll
        for (int k = 0; k < 4; ++k) {
            const int e = k * 32 + lane;
            _Float16 hv = (_Float16)0.0f;
            float cv = 0.0f;
            if (e < len) {                       // predicated: masked tail never read
                const int   gi = idx[base + e];
                const float xv = x[gi];
                const float yv = y[gi];
                hv = (_Float16)__expf(xv);       // |x| <~ 5.7 -> no f16 overflow
                cv = (yv == 1.0f) ? xv : 0.0f;
            }
            E[wave][rr][e] = hv;                 // zeros written for masked slots
            cacc += cv;
        }
        #pragma unroll
        for (int off = 16; off > 0; off >>= 1)   // exact f32 chosen-sum, deterministic
            cacc += __shfl_xor(cacc, off, 32);
        if (lane == rr) Ch[wave][rr] = cacc;
    }

    __syncthreads();

    // B operand = ones: D[m][n] = sum_k A[m][k]
    v16h bOnes;
    #pragma unroll
    for (int i = 0; i < 16; ++i) bOnes[i] = (_Float16)1.0f;

    // A-fragment per ISA 16-bit 16x32 layout:
    // lanes 0-15 (M=r): halves 0-7 = K 0-7,  halves 8-15 = K 16-23
    // lanes 16-31(M=r): halves 0-7 = K 8-15, halves 8-15 = K 24-31
    const int r  = lane & 15;
    const int hi = (lane >> 4) & 1;

    v8f acc = {};
    #pragma unroll
    for (int t = 0; t < 4; ++t) {                // K = 128 in 4 chained WMMAs
        const v8h* pl = (const v8h*)&E[wave][r][t * 32 + hi * 8];
        const v8h* ph = (const v8h*)&E[wave][r][t * 32 + 16 + hi * 8];
        v8h lo = *pl;
        v8h hh = *ph;
        v16h a = __builtin_shufflevector(lo, hh,
                 0, 1, 2, 3, 4, 5, 6, 7, 8, 9, 10, 11, 12, 13, 14, 15);
        acc = __builtin_amdgcn_wmma_f32_16x16x32_f16(
                  false, a, false, bOnes, (short)0, acc, false, false);
    }

    // D layout: lanes 0-15 hold rows 0-7 in acc[0..7]; lanes 16-31 hold rows 8-15
    float p = 0.0f;
    if (lane == 0) {
        #pragma unroll
        for (int j = 0; j < 8; ++j)
            if (rowBase + j < Brows) p += Ch[wave][j] - __logf(acc[j]);
    } else if (lane == 16) {
        #pragma unroll
        for (int j = 0; j < 8; ++j)
            if (rowBase + 8 + j < Brows) p += Ch[wave][8 + j] - __logf(acc[j]);
    }

    sdata[tid] = p;
    __syncthreads();
    #pragma unroll
    for (int s = 128; s > 0; s >>= 1) {
        if (tid < s) sdata[tid] += sdata[tid + s];
        __syncthreads();
    }
    if (tid == 0) ws[blockIdx.x] = sdata[0];
}

// Kernel 2: deterministic final reduction of block partials -> scalar loss.
__global__ __launch_bounds__(256) void mnl_reduce_kernel(
    const float* __restrict__ ws, float* __restrict__ out,
    int n, float negInvScale)
{
    __shared__ float sd[256];
    const int tid = threadIdx.x;
    float s = 0.0f;
    for (int i = tid; i < n; i += 256) s += ws[i];
    sd[tid] = s;
    __syncthreads();
    #pragma unroll
    for (int st = 128; st > 0; st >>= 1) {
        if (tid < st) sd[tid] += sd[tid + st];
        __syncthreads();
    }
    if (tid == 0) out[0] = sd[0] * negInvScale;   // -mean / ln(10)
}

extern "C" void kernel_launch(void* const* d_in, const int* in_sizes, int n_in,
                              void* d_out, int out_size, void* d_ws, size_t ws_size,
                              hipStream_t stream) {
    const float* x   = (const float*)d_in[0];
    const float* y   = (const float*)d_in[1];
    const int*   idx = (const int*)d_in[2];
    const int*   len = (const int*)d_in[3];
    float* out = (float*)d_out;
    float* ws  = (float*)d_ws;

    const int Brows  = in_sizes[3];              // 262144
    const int blocks = (Brows + 127) / 128;      // 128 rows per block

    mnl_main_kernel<<<blocks, 256, 0, stream>>>(x, y, idx, len, ws, Brows);

    const double LN10 = 2.302585092994045684;
    const float negInvScale = (float)(-1.0 / ((double)Brows * LN10));
    mnl_reduce_kernel<<<1, 256, 0, stream>>>(ws, out, blocks, negInvScale);
}
